// ConditionModulationBlock_6227702579490
// MI455X (gfx1250) — matrix operand order repaired and checked by reading, not compiled, not run on hardware
//
#include <hip/hip_runtime.h>
#include <hip/hip_bf16.h>
#include <math.h>

// ---------------- problem constants (from reference) ----------------
#define B_SZ        4
#define D_MODEL     1024
#define COND_DIM    768
#define COND_LEN    257
#define TOKEN_PNUM  4
#define TOKEN_PLEN  1024
#define PART_LEN    (COND_LEN + TOKEN_PLEN)          // 1281
#define L_TOTAL     (TOKEN_PNUM * PART_LEN)          // 5124
#define D_INNER     (2 * D_MODEL)                    // 2048
#define D_STATE     16
#define D_CONV      4
#define DT_RANK     64
#define DBC_DIM     (DT_RANK + 2 * D_STATE)          // 96
#define M_ROWS      (B_SZ * L_TOTAL)                 // 20496

typedef __attribute__((ext_vector_type(16))) _Float16 v16h;
typedef __attribute__((ext_vector_type(8)))  _Float16 v8h;
typedef __attribute__((ext_vector_type(8)))  float    v8f;

// =====================================================================
// f32 -> f16 convert (grid-stride)
// =====================================================================
__global__ __launch_bounds__(256) void k_cvt_f16(const float* __restrict__ src,
                                                 _Float16* __restrict__ dst,
                                                 size_t n) {
    size_t i = (size_t)blockIdx.x * blockDim.x + threadIdx.x;
    size_t stride = (size_t)gridDim.x * blockDim.x;
    for (; i < n; i += stride) dst[i] = (_Float16)src[i];
}

// strided slice convert: dtr[m,0:64] = (f16) dbc[m,0:64]  (dbc row = 96)
__global__ __launch_bounds__(256) void k_slice_dtr(const float* __restrict__ dbc,
                                                   _Float16* __restrict__ dtr,
                                                   size_t n) {
    size_t i = (size_t)blockIdx.x * blockDim.x + threadIdx.x;
    if (i >= n) return;
    size_t row = i / DT_RANK;
    size_t col = i - row * DT_RANK;
    dtr[i] = (_Float16)dbc[row * DBC_DIM + col];
}

// =====================================================================
// A-fragment loader: 16x32 f16, contiguous k order (same as B).
// lane<16: K = {0..15}; lane>=16: K = {16..31}, row = lane&15.
// =====================================================================
__device__ __forceinline__ v16h load_a_frag(const _Float16* __restrict__ Arow,
                                            int k0, int kOffLo, int kOffHi) {
    v8h alo = *(const v8h*)(Arow + k0 + kOffLo);
    v8h ahi = *(const v8h*)(Arow + k0 + kOffHi);
    return __builtin_shufflevector(alo, ahi,
            0,1,2,3,4,5,6,7,8,9,10,11,12,13,14,15);
}

// =====================================================================
// WMMA GEMM:  C[M,N] = A[M,K] (f16, row-major) @ W[N,K]^T (f16, row-major)
// Each wave computes a 32x64 tile: 2 A-frags x 4 B-frags -> 8 WMMAs/k-step.
// Each B fragment is loaded once and reused for both M sub-tiles
// (~21 FLOP/B at cache level vs ~13 for a 16x64 tile).
// mode: 0 = plain f32 store, 1 = +bias, 2 = softplus(x + bias)
// =====================================================================
__global__ __launch_bounds__(256) void k_wmma_gemm_nt(
    const _Float16* __restrict__ A, const _Float16* __restrict__ W,
    float* __restrict__ C, const float* __restrict__ bias,
    int M, int N, int K, int mode)
{
    const int lane   = threadIdx.x & 31;
    const int waveId = (blockIdx.x * (blockDim.x >> 5)) + (threadIdx.x >> 5);
    const int mGroups = (M + 31) >> 5;
    const int nGroups = (N + 63) >> 6;
    if (waveId >= mGroups * nGroups) return;

    const int mGroup = waveId % mGroups;
    const int nGroup = waveId / mGroups;
    const int m0     = mGroup * 32;
    const int nbase  = nGroup * 64;

    int n0[4];
    #pragma unroll
    for (int j = 0; j < 4; ++j) {
        int n = nbase + j * 16;
        n0[j] = (n > N - 16) ? (N - 16) : n;   // clamp (duplicate work, same data)
    }

    // --- A fragment addressing (row-clamped for ragged M) ---
    int rowA0 = m0 + (lane & 15);
    int rowA1 = rowA0 + 16;
    if (rowA0 > M - 1) rowA0 = M - 1;
    if (rowA1 > M - 1) rowA1 = M - 1;
    const _Float16* Arow0 = A + (size_t)rowA0 * K;
    const _Float16* Arow1 = A + (size_t)rowA1 * K;
    const int kOffLo = (lane < 16) ? 0  : 16;
    const int kOffHi = (lane < 16) ? 8  : 24;

    // --- B fragment addressing: W row-major => contiguous 16 halfs per lane ---
    const int kOffB = (lane < 16) ? 0 : 16;
    const _Float16* Wp[4];
    #pragma unroll
    for (int j = 0; j < 4; ++j)
        Wp[j] = W + (size_t)(n0[j] + (lane & 15)) * K + kOffB;

    v8f acc0[4] = {v8f{}, v8f{}, v8f{}, v8f{}};
    v8f acc1[4] = {v8f{}, v8f{}, v8f{}, v8f{}};

    for (int k0 = 0; k0 < K; k0 += 32) {
        v16h a0 = load_a_frag(Arow0, k0, kOffLo, kOffHi);
        v16h a1 = load_a_frag(Arow1, k0, kOffLo, kOffHi);
        #pragma unroll
        for (int j = 0; j < 4; ++j) {
            v16h b = *(const v16h*)(Wp[j] + k0);
            acc0[j] = __builtin_amdgcn_wmma_f32_16x16x32_f16(
                          false, a0, false, b, (short)0, acc0[j], false, false);
            acc1[j] = __builtin_amdgcn_wmma_f32_16x16x32_f16(
                          false, a1, false, b, (short)0, acc1[j], false, false);
        }
    }

    // --- epilogue: C/D layout = VGPR i -> row (i | i+8), col = lane&15 ---
    const int nlane = lane & 15;
    const int rbase = m0 + ((lane < 16) ? 0 : 8);
    #pragma unroll
    for (int j = 0; j < 4; ++j) {
        const int ncol = n0[j] + nlane;
        const float bv = (mode != 0) ? bias[ncol] : 0.0f;
        #pragma unroll
        for (int i = 0; i < 8; ++i) {
            int row = rbase + i;
            if (row < M) {
                float v = acc0[j][i] + bv;
                if (mode == 2) v = (v > 20.0f) ? v : log1pf(__expf(v));
                C[(size_t)row * N + ncol] = v;
            }
            int row1 = row + 16;
            if (row1 < M) {
                float v = acc1[j][i] + bv;
                if (mode == 2) v = (v > 20.0f) ? v : log1pf(__expf(v));
                C[(size_t)row1 * N + ncol] = v;
            }
        }
    }
}

// =====================================================================
// Assemble res (output #2) + LayerNorm -> hs (f16). One block per row.
// =====================================================================
__global__ __launch_bounds__(256) void k_assemble_ln(
    const float* __restrict__ hs_in, const float* __restrict__ res_in,
    const float* __restrict__ pre,
    const float* __restrict__ norm_w, const float* __restrict__ norm_b,
    float* __restrict__ res_out, _Float16* __restrict__ hs_h)
{
    __shared__ float s_red[256];
    const int r   = blockIdx.x;                 // 0 .. M_ROWS-1
    const int b   = r / L_TOTAL;
    const int t   = r - b * L_TOTAL;
    const int pos = t % PART_LEN;
    const int tid = threadIdx.x;

    float v[4];
    #pragma unroll
    for (int i = 0; i < 4; ++i) {
        int d = i * 256 + tid;
        float val;
        if (pos < COND_LEN)
            val = pre[((size_t)b * COND_LEN + pos) * D_MODEL + d];
        else
            val = hs_in[(size_t)r * D_MODEL + d] + res_in[(size_t)r * D_MODEL + d];
        v[i] = val;
        res_out[(size_t)r * D_MODEL + d] = val;
    }

    float s = v[0] + v[1] + v[2] + v[3];
    s_red[tid] = s; __syncthreads();
    for (int off = 128; off > 0; off >>= 1) {
        if (tid < off) s_red[tid] += s_red[tid + off];
        __syncthreads();
    }
    const float mean = s_red[0] * (1.0f / D_MODEL);
    __syncthreads();

    float q = 0.f;
    #pragma unroll
    for (int i = 0; i < 4; ++i) { float dd = v[i] - mean; q += dd * dd; }
    s_red[tid] = q; __syncthreads();
    for (int off = 128; off > 0; off >>= 1) {
        if (tid < off) s_red[tid] += s_red[tid + off];
        __syncthreads();
    }
    const float inv = rsqrtf(s_red[0] * (1.0f / D_MODEL) + 1e-5f);

    #pragma unroll
    for (int i = 0; i < 4; ++i) {
        int d = i * 256 + tid;
        float hn = (v[i] - mean) * inv * norm_w[d] + norm_b[d];
        hs_h[(size_t)r * D_MODEL + d] = (_Float16)hn;
    }
}

// =====================================================================
// Depthwise causal conv (D_CONV=4) + SiLU.  xc = xz[..., :D_INNER].
// =====================================================================
__global__ __launch_bounds__(256) void k_conv_silu(
    const float* __restrict__ xz, const float* __restrict__ conv_w,
    const float* __restrict__ conv_b,
    float* __restrict__ u, _Float16* __restrict__ u_h)
{
    size_t idx = (size_t)blockIdx.x * blockDim.x + threadIdx.x;
    const size_t total = (size_t)M_ROWS * D_INNER;
    if (idx >= total) return;
    const int c    = (int)(idx & (D_INNER - 1));
    const size_t rt = idx / D_INNER;            // b*L_TOTAL + t
    const int t    = (int)(rt % L_TOTAL);

    float acc = conv_b[c];
    #pragma unroll
    for (int j = 0; j < D_CONV; ++j) {
        int tt = t - (D_CONV - 1) + j;
        if (tt >= 0)
            acc += conv_w[c * D_CONV + j] *
                   xz[(rt - (D_CONV - 1) + j) * (2 * D_INNER) + c];
    }
    float su = acc / (1.0f + __expf(-acc));     // silu
    u[idx]   = su;
    u_h[idx] = (_Float16)su;
}

// =====================================================================
// Selective scan: one lane per (b, d) channel, 16 states in registers.
// Fuses + u*D, * silu(z), f16 output for the out-proj GEMM.
// =====================================================================
__global__ __launch_bounds__(256) void k_mamba_scan(
    const float* __restrict__ dt, const float* __restrict__ u,
    const float* __restrict__ dbc, const float* __restrict__ xz,
    const float* __restrict__ A_log, const float* __restrict__ D_param,
    _Float16* __restrict__ y_h)
{
    const int idx = blockIdx.x * blockDim.x + threadIdx.x;   // 0..B*D_INNER-1
    const int b = idx >> 11;            // / D_INNER
    const int d = idx & (D_INNER - 1);

    float Arow[D_STATE];
    #pragma unroll
    for (int s = 0; s < D_STATE; ++s) Arow[s] = -__expf(A_log[d * D_STATE + s]);
    const float Dp = D_param[d];

    float h[D_STATE];
    #pragma unroll
    for (int s = 0; s < D_STATE; ++s) h[s] = 0.f;

    const size_t base = (size_t)b * L_TOTAL;
    for (int t = 0; t < L_TOTAL; ++t) {
        const size_t row = base + t;
        const float dtv = dt[row * D_INNER + d];
        const float uv  = u[row * D_INNER + d];
        const float du  = dtv * uv;
        float y = 0.f;
        #pragma unroll
        for (int s = 0; s < D_STATE; ++s) {
            const float Bm = dbc[row * DBC_DIM + DT_RANK + s];
            const float Cm = dbc[row * DBC_DIM + DT_RANK + D_STATE + s];
            h[s] = __expf(dtv * Arow[s]) * h[s] + du * Bm;
            y += h[s] * Cm;
        }
        const float z  = xz[row * (2 * D_INNER) + D_INNER + d];
        const float sz = z / (1.0f + __expf(-z));
        y_h[row * D_INNER + d] = (_Float16)((y + uv * Dp) * sz);
    }
}

// =====================================================================
// host side
// =====================================================================
static inline int gemm_blocks(int M, int N) {
    int waves = ((M + 31) / 32) * ((N + 63) / 64);
    return (waves + 7) / 8;     // 8 waves per 256-thread block
}

extern "C" void kernel_launch(void* const* d_in, const int* in_sizes, int n_in,
                              void* d_out, int out_size, void* d_ws, size_t ws_size,
                              hipStream_t stream) {
    (void)in_sizes; (void)n_in; (void)out_size; (void)ws_size;
    const float* hidden   = (const float*)d_in[0];
    const float* residual = (const float*)d_in[1];
    const float* cond     = (const float*)d_in[2];
    /* d_in[3] = mod : unused by the reference */
    const float* cond_w   = (const float*)d_in[4];
    const float* cond_b   = (const float*)d_in[5];
    const float* norm_w   = (const float*)d_in[6];
    const float* norm_b   = (const float*)d_in[7];
    const float* in_proj_w  = (const float*)d_in[8];
    const float* conv_w     = (const float*)d_in[9];
    const float* conv_b     = (const float*)d_in[10];
    const float* x_proj_w   = (const float*)d_in[11];
    const float* dt_proj_w  = (const float*)d_in[12];
    const float* dt_proj_b  = (const float*)d_in[13];
    const float* A_log      = (const float*)d_in[14];
    const float* D_param    = (const float*)d_in[15];
    const float* out_proj_w = (const float*)d_in[16];

    float* out_y   = (float*)d_out;                                     // (B,L,D)
    float* out_res = (float*)d_out + (size_t)M_ROWS * D_MODEL;          // (B,L,D)

    // ---- workspace carve-up (256B aligned) ----
    char* ws = (char*)d_ws;
    size_t off = 0;
    auto alloc = [&](size_t bytes) -> void* {
        off = (off + 255) & ~(size_t)255;
        void* p = ws + off;
        off += bytes;
        return p;
    };
    _Float16* condw_h   = (_Float16*)alloc((size_t)D_MODEL * COND_DIM * 2);
    _Float16* inprojw_h = (_Float16*)alloc((size_t)2 * D_INNER * D_MODEL * 2);
    _Float16* xprojw_h  = (_Float16*)alloc((size_t)DBC_DIM * D_INNER * 2);
    _Float16* dtprojw_h = (_Float16*)alloc((size_t)D_INNER * DT_RANK * 2);
    _Float16* outprojw_h= (_Float16*)alloc((size_t)D_MODEL * D_INNER * 2);
    _Float16* cond_h    = (_Float16*)alloc((size_t)B_SZ * COND_LEN * COND_DIM * 2);
    float*    pre       = (float*)   alloc((size_t)B_SZ * COND_LEN * D_MODEL * 4);
    _Float16* hs_h      = (_Float16*)alloc((size_t)M_ROWS * D_MODEL * 2);
    float*    xz        = (float*)   alloc((size_t)M_ROWS * 2 * D_INNER * 4);
    float*    u_f       = (float*)   alloc((size_t)M_ROWS * D_INNER * 4);
    _Float16* u_h       = (_Float16*)alloc((size_t)M_ROWS * D_INNER * 2);
    float*    dbc       = (float*)   alloc((size_t)M_ROWS * DBC_DIM * 4);
    _Float16* dtr_h     = (_Float16*)alloc((size_t)M_ROWS * DT_RANK * 2);
    float*    dt_f      = (float*)   alloc((size_t)M_ROWS * D_INNER * 4);
    _Float16* y_h       = (_Float16*)alloc((size_t)M_ROWS * D_INNER * 2);

    auto cvt = [&](const float* s, _Float16* dst, size_t n) {
        int blocks = (int)((n + 255) / 256);
        if (blocks > 65535) blocks = 65535;
        k_cvt_f16<<<blocks, 256, 0, stream>>>(s, dst, n);
    };

    // 1) weight / activation conversions to f16
    cvt(cond_w,     condw_h,    (size_t)D_MODEL * COND_DIM);
    cvt(in_proj_w,  inprojw_h,  (size_t)2 * D_INNER * D_MODEL);
    cvt(x_proj_w,   xprojw_h,   (size_t)DBC_DIM * D_INNER);
    cvt(dt_proj_w,  dtprojw_h,  (size_t)D_INNER * DT_RANK);
    cvt(out_proj_w, outprojw_h, (size_t)D_MODEL * D_INNER);
    cvt(cond,       cond_h,     (size_t)B_SZ * COND_LEN * COND_DIM);

    // 2) pre = cond @ cond_w^T + cond_b       (M=1028, N=1024, K=768)
    {
        const int M = B_SZ * COND_LEN, N = D_MODEL, K = COND_DIM;
        k_wmma_gemm_nt<<<gemm_blocks(M, N), 256, 0, stream>>>(
            cond_h, condw_h, pre, cond_b, M, N, K, /*mode=*/1);
    }

    // 3) res (output #2) + layernorm -> hs_h
    k_assemble_ln<<<M_ROWS, 256, 0, stream>>>(
        hidden, residual, pre, norm_w, norm_b, out_res, hs_h);

    // 4) xz = hs @ in_proj_w^T                (M=20496, N=4096, K=1024)
    k_wmma_gemm_nt<<<gemm_blocks(M_ROWS, 2 * D_INNER), 256, 0, stream>>>(
        hs_h, inprojw_h, xz, nullptr, M_ROWS, 2 * D_INNER, D_MODEL, 0);

    // 5) depthwise causal conv + silu -> u
    {
        size_t total = (size_t)M_ROWS * D_INNER;
        k_conv_silu<<<(int)((total + 255) / 256), 256, 0, stream>>>(
            xz, conv_w, conv_b, u_f, u_h);
    }

    // 6) dbc = u @ x_proj_w^T                 (M=20496, N=96, K=2048)
    k_wmma_gemm_nt<<<gemm_blocks(M_ROWS, DBC_DIM), 256, 0, stream>>>(
        u_h, xprojw_h, dbc, nullptr, M_ROWS, DBC_DIM, D_INNER, 0);

    // 7) dt_r slice -> f16
    {
        size_t n = (size_t)M_ROWS * DT_RANK;
        k_slice_dtr<<<(int)((n + 255) / 256), 256, 0, stream>>>(dbc, dtr_h, n);
    }

    // 8) dt = softplus(dt_r @ dt_proj_w^T + b)  (M=20496, N=2048, K=64)
    k_wmma_gemm_nt<<<gemm_blocks(M_ROWS, D_INNER), 256, 0, stream>>>(
        dtr_h, dtprojw_h, dt_f, dt_proj_b, M_ROWS, D_INNER, DT_RANK, /*mode=*/2);

    // 9) selective scan -> y_h (fused +u*D and *silu(z))
    k_mamba_scan<<<(B_SZ * D_INNER) / 256, 256, 0, stream>>>(
        dt_f, u_f, dbc, xz, A_log, D_param, y_h);

    // 10) out = y @ out_proj_w^T              (M=20496, N=1024, K=2048)
    k_wmma_gemm_nt<<<gemm_blocks(M_ROWS, D_MODEL), 256, 0, stream>>>(
        y_h, outprojw_h, out_y, nullptr, M_ROWS, D_MODEL, D_INNER, 0);
}
